// AttentiveNonLocalBlock2D_75076028334857
// MI455X (gfx1250) — matrix operand order repaired and checked by reading, not compiled, not run on hardware
//
#include <hip/hip_runtime.h>
#include <hip/hip_bf16.h>
#include <cstddef>

// ---------------------------------------------------------------------------
// AttentiveNonLocalBlock2D for MI455X (gfx1250, wave32, WMMA)
//
// Shapes: x [1,64,96,96] f32; N = 9216 tokens, C = 64, Ci = 32.
// Heavy part = non-local attention: f = theta^T phi (N x N), softmax over n
// (axis 1 == per-COLUMN m), y2 = f g^T.  f is 340 MB fp32 (> 192 MB L2), so
// we never materialize it: pass 1 = online per-column (max, sumexp) stats,
// pass 2 = recompute score tiles, scale, and chain into the AV WMMA.
// Both passes use v_wmma_f32_16x16x32_f16 (K=32 == Ci, one WMMA per tile).
// ---------------------------------------------------------------------------

typedef __attribute__((ext_vector_type(16))) _Float16 v16h;
typedef __attribute__((ext_vector_type(8)))  _Float16 v8h;
typedef __attribute__((ext_vector_type(8)))  float    v8f;

union V16H { v16h v; v8h h[2]; };

#define WMMA_F16(A, B, C) \
  __builtin_amdgcn_wmma_f32_16x16x32_f16(false, (A), false, (B), (short)0, (C), false, false)

static constexpr int Ntok = 9216;   // 96*96
static constexpr int Cfull = 64;
static constexpr int Ci = 32;

// ---------------- workspace layout (float offsets; halves region after) ----
static constexpr size_t OFF_Y1  = 0;        // 64*48*48 = 147456 f
static constexpr size_t OFF_Y2C = 147456;   // 64*24*24 =  36864 f
static constexpr size_t OFF_Y3  = 184320;   // 64*12*12 =   9216 f
static constexpr size_t OFF_XG  = 193536;   // 64*9216  = 589824 f (gated x)
static constexpr size_t OFF_Y2A = 783360;   // 32*9216  = 294912 f (attention out)
static constexpr size_t OFF_M   = 1078272;  // 9216 f  (per-column max)
static constexpr size_t OFF_Z   = 1087488;  // 9216 f  (per-column sumexp)
static constexpr size_t OFF_HLF = 1096704;  // halves start here (byte-off 4386816, 256B aligned)
static constexpr size_t H_THETA = 0;        // thetaT [n][32] f16
static constexpr size_t H_PHI   = 294912;   // phiT   [n][32] f16
static constexpr size_t H_G     = 589824;   // g      [c][9216] f16
// total ws bytes: 4386816 + 3*589824 = 6,156,288 (~6.2 MB)

// ---------------------------------------------------------------------------
// 3x3 stride-2 pad-1 conv, optional leaky-relu(0.2)
// ---------------------------------------------------------------------------
__global__ void conv3x3_s2_kernel(const float* __restrict__ in,
                                  const float* __restrict__ w,
                                  const float* __restrict__ bias,
                                  float* __restrict__ out,
                                  int C, int Hin, int Hout, int do_lrelu) {
  int id = blockIdx.x * blockDim.x + threadIdx.x;
  int total = C * Hout * Hout;
  if (id >= total) return;
  int ox = id % Hout;
  int oy = (id / Hout) % Hout;
  int o  = id / (Hout * Hout);
  float acc = bias[o];
  for (int ci = 0; ci < C; ++ci) {
    const float* ip = in + (size_t)ci * Hin * Hin;
    const float* wp = w + ((size_t)o * C + ci) * 9;
#pragma unroll
    for (int ky = 0; ky < 3; ++ky) {
      int iy = oy * 2 - 1 + ky;
      if ((unsigned)iy >= (unsigned)Hin) continue;
#pragma unroll
      for (int kx = 0; kx < 3; ++kx) {
        int ix = ox * 2 - 1 + kx;
        if ((unsigned)ix >= (unsigned)Hin) continue;
        acc += ip[iy * Hin + ix] * wp[ky * 3 + kx];
      }
    }
  }
  if (do_lrelu) acc = (acc >= 0.f) ? acc : 0.2f * acc;
  out[id] = acc;
}

// ---------------------------------------------------------------------------
// bilinear x8 upsample (half-pixel centers) of y3 [64,12,12] -> sigmoid gate
// xg = sigmoid(up(y3)) * x      [64,96,96]
// ---------------------------------------------------------------------------
__global__ void gate_kernel(const float* __restrict__ x,
                            const float* __restrict__ y3,
                            float* __restrict__ xg) {
  int id = blockIdx.x * blockDim.x + threadIdx.x;
  if (id >= Cfull * Ntok) return;
  int ox = id % 96;
  int oy = (id / 96) % 96;
  int c  = id / (96 * 96);
  float fy = (oy + 0.5f) * 0.125f - 0.5f;
  float fx = (ox + 0.5f) * 0.125f - 0.5f;
  int y0 = (int)floorf(fy);  float ty = fy - (float)y0;
  int x0 = (int)floorf(fx);  float tx = fx - (float)x0;
  int y0c = min(max(y0, 0), 11), y1c = min(max(y0 + 1, 0), 11);
  int x0c = min(max(x0, 0), 11), x1c = min(max(x0 + 1, 0), 11);
  const float* p = y3 + (size_t)c * 144;
  float v = (1.f - ty) * ((1.f - tx) * p[y0c * 12 + x0c] + tx * p[y0c * 12 + x1c]) +
            ty         * ((1.f - tx) * p[y1c * 12 + x0c] + tx * p[y1c * 12 + x1c]);
  float s = 1.f / (1.f + __expf(-v));
  xg[id] = s * x[id];
}

// ---------------------------------------------------------------------------
// 1x1 projections theta/phi/g (64 -> 32), emitted in WMMA-friendly f16 layouts
//   thetaT [n][32], phiT [n][32], g [c][9216]
// ---------------------------------------------------------------------------
__global__ void proj_kernel(const float* __restrict__ xg,
                            const float* __restrict__ thw, const float* __restrict__ thb,
                            const float* __restrict__ phw, const float* __restrict__ phb,
                            const float* __restrict__ gw,  const float* __restrict__ gb,
                            _Float16* __restrict__ thetaT,
                            _Float16* __restrict__ phiT,
                            _Float16* __restrict__ gh) {
  int n = blockIdx.x * blockDim.x + threadIdx.x;
  if (n >= Ntok) return;
  int c = blockIdx.y;      // 0..31
  int which = blockIdx.z;  // 0=theta 1=phi 2=g
  const float* w; const float* b;
  if (which == 0)      { w = thw; b = thb; }
  else if (which == 1) { w = phw; b = phb; }
  else                 { w = gw;  b = gb;  }
  float acc = b[c];
  const float* wr = w + (size_t)c * Cfull;
#pragma unroll 8
  for (int k = 0; k < Cfull; ++k) acc += wr[k] * xg[(size_t)k * Ntok + n];
  if (which == 0)      thetaT[(size_t)n * 32 + c] = (_Float16)acc;
  else if (which == 1) phiT[(size_t)n * 32 + c]   = (_Float16)acc;
  else                 gh[(size_t)c * Ntok + n]   = (_Float16)acc;
}

// ---------------------------------------------------------------------------
// Pass 1: per-column softmax stats (softmax over n for each column m).
// One wave owns 16 columns (B operand fixed) and sweeps all n-tiles:
//   S-tile = WMMA( thetaT n-tile [16n x 32c], phiT m-tile [32c x 16m] )
// C/D layout: lane holds ONE column (N = lane%16), 8 rows; fold with the
// xor-16 partner lane, then classic online (max, sumexp) update.
// ---------------------------------------------------------------------------
__global__ void __launch_bounds__(256)
attn_stats_kernel(const _Float16* __restrict__ thetaT,
                  const _Float16* __restrict__ phiT,
                  float* __restrict__ Mst, float* __restrict__ Zst) {
  int wave = (blockIdx.x * blockDim.x + threadIdx.x) >> 5;
  int lane = threadIdx.x & 31;
  int ln = lane & 15;
  bool hi = lane >= 16;
  int m0 = wave * 16;

  // B operand: phi [K=c, N=m]; lane = column, halves = c 0..15 (lo) / 16..31 (hi)
  V16H B;
  const _Float16* prow = phiT + (size_t)(m0 + ln) * 32 + (hi ? 16 : 0);
  B.h[0] = *(const v8h*)(prow);
  B.h[1] = *(const v8h*)(prow + 8);

  float runM = -3.4e38f, runZ = 0.f;
  for (int n0 = 0; n0 < Ntok; n0 += 16) {
    // A operand: theta^T [M=n, K=c]; lane = row, halves = c 0..15 (lo) / 16..31 (hi)
    const _Float16* arow = thetaT + (size_t)(n0 + ln) * 32 + (hi ? 16 : 0);
    V16H A;
    A.h[0] = *(const v8h*)(arow);
    A.h[1] = *(const v8h*)(arow + 8);
    v8f cz = {};
    v8f s = WMMA_F16(A.v, B.v, cz);

    float tmax = s[0];
#pragma unroll
    for (int r = 1; r < 8; ++r) tmax = fmaxf(tmax, s[r]);
    tmax = fmaxf(tmax, __shfl_xor(tmax, 16, 32));     // fold 16-row halves
    float newM = fmaxf(runM, tmax);
    float part = 0.f;
#pragma unroll
    for (int r = 0; r < 8; ++r) part += __expf(s[r] - newM);
    part += __shfl_xor(part, 16, 32);                 // partner used same newM
    runZ = runZ * __expf(runM - newM) + part;
    runM = newM;
  }
  if (lane < 16) {
    Mst[m0 + ln] = runM;
    Zst[m0 + ln] = runZ;
  }
}

// ---------------------------------------------------------------------------
// Pass 2: y2[c,n] = sum_m P[n,m] g[c,m],  P = exp(S - M[m]) / Z[m].
// One wave owns 16 n (B'' = theta tile fixed), sweeps m in chunks of 32:
//   T-tiles = WMMA( phiT m-tiles, theta n-tile )  -> T[m,n] (transposed score)
//   scale rows by column stats, cvt f16, each lane packs its own rows of the two
//   T C/D tiles into one B-operand [K=32m x 16n], then
//   acc += WMMA( g c-tile [16c x 32m], P [32m x 16n] )   (x2 c-tiles)
// ---------------------------------------------------------------------------
__global__ void __launch_bounds__(256)
attn_av_kernel(const _Float16* __restrict__ thetaT,
               const _Float16* __restrict__ phiT,
               const _Float16* __restrict__ gh,
               const float* __restrict__ Mst, const float* __restrict__ Zst,
               float* __restrict__ y2) {
  int wave = (blockIdx.x * blockDim.x + threadIdx.x) >> 5;
  int lane = threadIdx.x & 31;
  int ln = lane & 15;
  bool hi = lane >= 16;
  int n0 = wave * 16;

  // B'' operand: theta [K=c, N=n]
  V16H Bn;
  const _Float16* trow = thetaT + (size_t)(n0 + ln) * 32 + (hi ? 8 : 0);
  Bn.h[0] = *(const v8h*)(trow);
  Bn.h[1] = *(const v8h*)(trow + 16);

  v8f acc0 = {}, acc1 = {};
  for (int m0 = 0; m0 < Ntok; m0 += 32) {
    // A'' operands: phi^T rows m0..m0+15 and m0+16..m0+31
    V16H A1, A2;
    const _Float16* p1 = phiT + (size_t)(m0 + ln) * 32 + (hi ? 8 : 0);
    A1.h[0] = *(const v8h*)(p1);
    A1.h[1] = *(const v8h*)(p1 + 16);
    const _Float16* p2 = phiT + (size_t)(m0 + 16 + ln) * 32 + (hi ? 8 : 0);
    A2.h[0] = *(const v8h*)(p2);
    A2.h[1] = *(const v8h*)(p2 + 16);
    v8f cz = {};
    v8f t1 = WMMA_F16(A1.v, Bn.v, cz);   // T[m0+0..15 , n]
    v8f t2 = WMMA_F16(A2.v, Bn.v, cz);   // T[m0+16..31, n]

    // softmax-scale rows and repack into B-operand P [K=32m x N=16n]
    V16H P;
#pragma unroll
    for (int r = 0; r < 8; ++r) {
      int mr1 = m0 + r + (hi ? 8 : 0);       // row of t1[r]
      int mr2 = mr1 + 16;                    // row of t2[r]
      float e1 = __expf(t1[r] - Mst[mr1]) / Zst[mr1];
      float e2 = __expf(t2[r] - Mst[mr2]) / Zst[mr2];
      float o1 = __shfl_xor(e1, 16, 32);     // partner half's rows
      float o2 = __shfl_xor(e2, 16, 32);
      // lo lanes need K = m0+{0-7,16-23} (own t1 + own t2)
      // hi lanes need K = m0+{8-15,24-31} (own t1 + own t2)
      P.v[r]     = (_Float16)e1;
      P.v[8 + r] = (_Float16)e2;
    }

    // A' operands: g c-tiles [M=c, K=m], halves = m {0-7,16-23}/{8-15,24-31}
    V16H G0, G1;
    const _Float16* g0 = gh + (size_t)ln * Ntok + m0 + (hi ? 8 : 0);
    G0.h[0] = *(const v8h*)(g0);
    G0.h[1] = *(const v8h*)(g0 + 16);
    const _Float16* g1 = gh + (size_t)(16 + ln) * Ntok + m0 + (hi ? 8 : 0);
    G1.h[0] = *(const v8h*)(g1);
    G1.h[1] = *(const v8h*)(g1 + 16);
    acc0 = WMMA_F16(G0.v, P.v, acc0);    // c 0..15
    acc1 = WMMA_F16(G1.v, P.v, acc1);    // c 16..31
  }

  // D layout: lane = column n, rows c split across lane halves
#pragma unroll
  for (int r = 0; r < 8; ++r) {
    int c = r + (hi ? 8 : 0);
    y2[(size_t)c * Ntok + n0 + ln]        = acc0[r];
    y2[(size_t)(c + 16) * Ntok + n0 + ln] = acc1[r];
  }
}

// ---------------------------------------------------------------------------
// final: out[o,n] = sum_c W_w[o,c] * y2[c,n] + W_b[o] + xg[o,n]
// ---------------------------------------------------------------------------
__global__ void final_kernel(const float* __restrict__ y2,
                             const float* __restrict__ Ww,
                             const float* __restrict__ Wb,
                             const float* __restrict__ xg,
                             float* __restrict__ out) {
  int id = blockIdx.x * blockDim.x + threadIdx.x;
  if (id >= Cfull * Ntok) return;
  int n = id % Ntok;
  int o = id / Ntok;
  float acc = Wb[o];
  const float* wr = Ww + (size_t)o * Ci;
#pragma unroll
  for (int c = 0; c < Ci; ++c) acc += wr[c] * y2[(size_t)c * Ntok + n];
  out[id] = acc + xg[id];
}

// ---------------------------------------------------------------------------
extern "C" void kernel_launch(void* const* d_in, const int* in_sizes, int n_in,
                              void* d_out, int out_size, void* d_ws, size_t ws_size,
                              hipStream_t stream) {
  (void)in_sizes; (void)n_in; (void)out_size; (void)ws_size;
  const float* x    = (const float*)d_in[0];
  const float* d1w  = (const float*)d_in[1];
  const float* d1b  = (const float*)d_in[2];
  const float* d2w  = (const float*)d_in[3];
  const float* d2b  = (const float*)d_in[4];
  const float* d3w  = (const float*)d_in[5];
  const float* d3b  = (const float*)d_in[6];
  const float* gw   = (const float*)d_in[7];
  const float* gb   = (const float*)d_in[8];
  const float* thw  = (const float*)d_in[9];
  const float* thb  = (const float*)d_in[10];
  const float* phw  = (const float*)d_in[11];
  const float* phb  = (const float*)d_in[12];
  const float* Ww   = (const float*)d_in[13];
  const float* Wb   = (const float*)d_in[14];
  float* out = (float*)d_out;

  float* wsf = (float*)d_ws;
  float* y1   = wsf + OFF_Y1;
  float* y2c  = wsf + OFF_Y2C;
  float* y3   = wsf + OFF_Y3;
  float* xg   = wsf + OFF_XG;
  float* y2a  = wsf + OFF_Y2A;
  float* Mst  = wsf + OFF_M;
  float* Zst  = wsf + OFF_Z;
  _Float16* hbase  = (_Float16*)(wsf + OFF_HLF);
  _Float16* thetaT = hbase + H_THETA;
  _Float16* phiT   = hbase + H_PHI;
  _Float16* ghalf  = hbase + H_G;

  // gating branch: 3x stride-2 convs
  conv3x3_s2_kernel<<<(64 * 48 * 48 + 255) / 256, 256, 0, stream>>>(x,   d1w, d1b, y1,  64, 96, 48, 1);
  conv3x3_s2_kernel<<<(64 * 24 * 24 + 255) / 256, 256, 0, stream>>>(y1,  d2w, d2b, y2c, 64, 48, 24, 1);
  conv3x3_s2_kernel<<<(64 * 12 * 12 + 255) / 256, 256, 0, stream>>>(y2c, d3w, d3b, y3,  64, 24, 12, 0);
  // bilinear x8 upsample + sigmoid gate
  gate_kernel<<<(Cfull * Ntok + 255) / 256, 256, 0, stream>>>(x, y3, xg);
  // 1x1 projections -> f16 WMMA layouts
  proj_kernel<<<dim3(Ntok / 256, 32, 3), 256, 0, stream>>>(
      xg, thw, thb, phw, phb, gw, gb, thetaT, phiT, ghalf);
  // flash-style attention: 576 waves each pass (16 cols / 16 rows per wave)
  attn_stats_kernel<<<72, 256, 0, stream>>>(thetaT, phiT, Mst, Zst);
  attn_av_kernel<<<72, 256, 0, stream>>>(thetaT, phiT, ghalf, Mst, Zst, y2a);
  // output 1x1 conv + gated residual
  final_kernel<<<(Cfull * Ntok + 255) / 256, 256, 0, stream>>>(y2a, Ww, Wb, xg, out);
}
